// Swin_Transformer_3D_21809843929156
// MI455X (gfx1250) — matrix operand order repaired and checked by reading, not compiled, not run on hardware
//
#include <hip/hip_runtime.h>
#include <cstdint>

typedef __attribute__((ext_vector_type(16))) _Float16 v16h;
typedef __attribute__((ext_vector_type(8)))  float    v8f;
typedef int v4i __attribute__((vector_size(16)));

#define WMMA16(a, b, c) \
  __builtin_amdgcn_wmma_f32_16x16x32_f16(false, (a), false, (b), (short)0, (c), false, false)

// gfx1250 async global->LDS copy path (ASYNCcnt), guarded so absence still compiles.
#if defined(__has_builtin)
#if __has_builtin(__builtin_amdgcn_global_load_async_to_lds_b128) && \
    __has_builtin(__builtin_amdgcn_s_wait_asynccnt)
#define HAVE_ASYNC_LDS 1
#endif
#endif
#ifndef HAVE_ASYNC_LDS
#define HAVE_ASYNC_LDS 0
#endif
#define AS1 __attribute__((address_space(1)))
#define AS3 __attribute__((address_space(3)))

// ---------------------------------------------------------------- helpers
__device__ __forceinline__ float wave_sum32(float v) {
#pragma unroll
  for (int o = 16; o >= 1; o >>= 1) v += __shfl_xor(v, o, 32);
  return v;
}

// ---------------------------------------------------------------- LayerNorm (block = C threads)
__global__ void layernorm_k(const float* __restrict__ in, float* __restrict__ out,
                            const float* __restrict__ g, const float* __restrict__ b, int C) {
  __shared__ float partial[12];
  __shared__ float stats[2];
  const int t = blockIdx.x;
  const int c = threadIdx.x;
  const int lane = c & 31, wv = c >> 5;
  const int nw = blockDim.x >> 5;
  float v = in[(size_t)t * C + c];
  float s = wave_sum32(v);
  if (lane == 0) partial[wv] = s;
  __syncthreads();
  if (c == 0) {
    float tot = 0.f;
    for (int i = 0; i < nw; ++i) tot += partial[i];
    stats[0] = tot / (float)C;
  }
  __syncthreads();
  float d = v - stats[0];
  s = wave_sum32(d * d);
  if (lane == 0) partial[wv] = s;
  __syncthreads();
  if (c == 0) {
    float tot = 0.f;
    for (int i = 0; i < nw; ++i) tot += partial[i];
    stats[1] = rsqrtf(tot / (float)C + 1e-5f);
  }
  __syncthreads();
  out[(size_t)t * C + c] = d * stats[1] * g[c] + b[c];
}

// ---------------------------------------------------------------- patch embed (conv 4x4x4 s4) + LN
__global__ __launch_bounds__(96) void patch_embed_k(
    const float* __restrict__ x, const float* __restrict__ pw, const float* __restrict__ pbias,
    const float* __restrict__ ng, const float* __restrict__ nb, float* __restrict__ out) {
  __shared__ float patch[64];
  __shared__ float partial[3];
  __shared__ float stats[2];
  int r = blockIdx.x;  // token over (2,8,56,56)
  const int w = r % 56; r /= 56;
  const int h = r % 56; r /= 56;
  const int d = r % 8;  r /= 8;
  const int b = r;
  const int c = threadIdx.x;
  const int lane = c & 31, wv = c >> 5;
  if (c < 64) {
    const int i = c >> 4, j = (c >> 2) & 3, k = c & 3;
    patch[c] = x[(((size_t)b * 32 + 4 * d + i) * 224 + 4 * h + j) * 224 + 4 * w + k];
  }
  __syncthreads();
  float acc = pbias[c];
#pragma unroll
  for (int e = 0; e < 64; ++e) acc += patch[e] * pw[c * 64 + e];
  float s = wave_sum32(acc);
  if (lane == 0) partial[wv] = s;
  __syncthreads();
  if (c == 0) stats[0] = (partial[0] + partial[1] + partial[2]) / 96.f;
  __syncthreads();
  float dv = acc - stats[0];
  s = wave_sum32(dv * dv);
  if (lane == 0) partial[wv] = s;
  __syncthreads();
  if (c == 0) stats[1] = rsqrtf((partial[0] + partial[1] + partial[2]) / 96.f + 1e-5f);
  __syncthreads();
  out[(size_t)blockIdx.x * 96 + c] = dv * stats[1] * ng[c] + nb[c];
}

// ---------------------------------------------------------------- window partition / reverse (+roll)
__device__ __forceinline__ size_t win_grid_index(size_t i, int Dg, int Hg, int Wg, int C,
                                                 int sd, int sh, int sw) {
  const int nWh = Hg / 7, nWw = Wg / 7;
  const int nWd = Dg / 4;
  const int c = (int)(i % C);
  size_t e = i / C;
  const int m = (int)(e % 196);
  size_t wdx = e / 196;
  const int ww = (int)(wdx % nWw); wdx /= nWw;
  const int wh = (int)(wdx % nWh); wdx /= nWh;
  const int wd = (int)(wdx % nWd);
  const int b  = (int)(wdx / nWd);
  const int dd = m / 49, rem = m % 49, hh = rem / 7, w7 = rem % 7;
  const int gd = (wd * 4 + dd + sd) % Dg;
  const int gh = (wh * 7 + hh + sh) % Hg;
  const int gw = (ww * 7 + w7 + sw) % Wg;
  return ((((size_t)b * Dg + gd) * Hg + gh) * Wg + gw) * (size_t)C + c;
}

__global__ void win_partition_k(const float* __restrict__ H, float* __restrict__ Wb,
                                int Dg, int Hg, int Wg, int C, int sd, int sh, int sw,
                                size_t total) {
  for (size_t i = (size_t)blockIdx.x * blockDim.x + threadIdx.x; i < total;
       i += (size_t)gridDim.x * blockDim.x) {
    Wb[i] = H[win_grid_index(i, Dg, Hg, Wg, C, sd, sh, sw)];
  }
}

__global__ void win_reverse_add_k(const float* __restrict__ Pr, float* __restrict__ X,
                                  int Dg, int Hg, int Wg, int C, int sd, int sh, int sw,
                                  size_t total) {
  for (size_t i = (size_t)blockIdx.x * blockDim.x + threadIdx.x; i < total;
       i += (size_t)gridDim.x * blockDim.x) {
    X[win_grid_index(i, Dg, Hg, Wg, C, sd, sh, sw)] += Pr[i];
  }
}

// ---------------------------------------------------------------- generic WMMA GEMM
// out = act(A @ W^T + bias) (+res).  A: MxK f32, W: NxK f32.  ACT: 0 none, 1 exact GELU.
// 128x64 block tile; 8 waves each own a 32x32 macro-tile => 4 WMMA / wave / K-step.
// K is always a multiple of 32 here. Staging uses GLOBAL_LOAD_ASYNC_TO_LDS_B128 when available;
// out-of-range rows/cols are simply skipped (their LDS garbage only feeds outputs the
// bounds-checked epilogue never stores, and WMMA raises no exceptions).
template <int ACT, bool RES>
__global__ __launch_bounds__(256) void gemm_wmma_k(
    const float* __restrict__ A, const float* __restrict__ W, const float* __restrict__ bias,
    const float* __restrict__ res, float* __restrict__ out, int M, int N, int K) {
  __shared__ float Asf[128][32];
  __shared__ float Bsf[64][32];
  const int tid = threadIdx.x;
  const int lane = tid & 31;
  const int wave = tid >> 5;
  const int wm = wave >> 1;  // 0..3
  const int wn = wave & 1;   // 0..1
  const int m0 = blockIdx.y * 128;
  const int n0 = blockIdx.x * 64;
  const int lm = lane & 15;
  const int hi = lane >> 4;

  v8f c00 = {}, c01 = {}, c10 = {}, c11 = {};
  for (int kk = 0; kk < K; kk += 32) {
#if HAVE_ASYNC_LDS
    for (int e = tid * 4; e < 128 * 32; e += 256 * 4) {
      const int r = e >> 5, k = e & 31;
      const int gm = m0 + r;
      if (gm < M)
        __builtin_amdgcn_global_load_async_to_lds_b128(
            (AS1 v4i*)(A + (size_t)gm * K + kk + k), (AS3 v4i*)&Asf[r][k], 0, 0);
    }
    for (int e = tid * 4; e < 64 * 32; e += 256 * 4) {
      const int r = e >> 5, k = e & 31;
      const int gn = n0 + r;
      if (gn < N)
        __builtin_amdgcn_global_load_async_to_lds_b128(
            (AS1 v4i*)(W + (size_t)gn * K + kk + k), (AS3 v4i*)&Bsf[r][k], 0, 0);
    }
#else
    for (int e = tid; e < 128 * 32; e += 256) {
      const int r = e >> 5, k = e & 31;
      const int gm = m0 + r;
      Asf[r][k] = (gm < M) ? A[(size_t)gm * K + kk + k] : 0.f;
    }
    for (int e = tid; e < 64 * 32; e += 256) {
      const int r = e >> 5, k = e & 31;
      const int gn = n0 + r;
      Bsf[r][k] = (gn < N) ? W[(size_t)gn * K + kk + k] : 0.f;
    }
#endif
    if (kk + 32 < K) {
      int prow = m0 + (tid >> 1);
      if (prow >= M) prow = M - 1;
      __builtin_prefetch(A + (size_t)prow * K + kk + 32, 0, 1);  // global_prefetch_b8
    }
#if HAVE_ASYNC_LDS
    __builtin_amdgcn_s_wait_asynccnt(0);
#endif
    __syncthreads();
    v16h a0, a1, b0, b1;
#pragma unroll
    for (int i = 0; i < 16; ++i) {
      const int ka = (hi << 4) | i;
      a0[i] = (_Float16)Asf[wm * 32 + lm][ka];
      a1[i] = (_Float16)Asf[wm * 32 + 16 + lm][ka];
      const int kb = (hi << 4) | i;
      b0[i] = (_Float16)Bsf[wn * 32 + lm][kb];
      b1[i] = (_Float16)Bsf[wn * 32 + 16 + lm][kb];
    }
    c00 = WMMA16(a0, b0, c00);
    c01 = WMMA16(a0, b1, c01);
    c10 = WMMA16(a1, b0, c10);
    c11 = WMMA16(a1, b1, c11);
    __syncthreads();
  }
  auto epi = [&](const v8f& cc, int ti, int tj) {
#pragma unroll
    for (int r = 0; r < 8; ++r) {
      const int m = m0 + wm * 32 + ti * 16 + r + 8 * hi;
      const int n = n0 + wn * 32 + tj * 16 + lm;
      if (m < M && n < N) {
        float v = cc[r];
        if (bias) v += bias[n];
        if (ACT == 1) v = 0.5f * v * (1.f + erff(v * 0.70710678118f));
        const size_t idx = (size_t)m * N + n;
        if (RES) v += res[idx];
        out[idx] = v;
      }
    }
  };
  epi(c00, 0, 0);
  epi(c01, 0, 1);
  epi(c10, 1, 0);
  epi(c11, 1, 1);
}

// ---------------------------------------------------------------- fused window attention (N=196, hd=32)
#define NTOK 196
#define NPAD 224
__global__ __launch_bounds__(64) void win_attn_k(
    const float* __restrict__ qkv, const float* __restrict__ rpb, float* __restrict__ out,
    int heads, int C, int Dg, int Hg, int Wg, int shifted) {
  __shared__ _Float16 Qs[NPAD][32];
  __shared__ _Float16 Ks[NPAD][32];
  __shared__ _Float16 Vs[NPAD][32];
  __shared__ _Float16 Ps[2][16][NPAD];
  __shared__ int cnt[NTOK];

  const int w = blockIdx.x;
  const int hI = blockIdx.y;
  const int tid = threadIdx.x;
  const int lane = tid & 31;
  const int wave = tid >> 5;
  const int C3 = 3 * C;
  const size_t base = (size_t)w * NTOK * C3;

  for (int e = tid; e < NPAD * 32; e += 64) {
    const int row = e >> 5, d = e & 31;
    _Float16 q = (_Float16)0.f, k = (_Float16)0.f, v = (_Float16)0.f;
    if (row < NTOK) {
      const float* p = qkv + base + (size_t)row * C3 + hI * 32 + d;
      q = (_Float16)(p[0] * 0.17677669529663687f);  // 1/sqrt(32)
      k = (_Float16)p[C];
      v = (_Float16)p[2 * C];
    }
    Qs[row][d] = q; Ks[row][d] = k; Vs[row][d] = v;
  }
  if (shifted) {
    const int nWh = Hg / 7, nWw = Wg / 7, nWd = Dg / 4;
    const int wl = w % (nWd * nWh * nWw);
    const int www = wl % nWw, whh = (wl / nWw) % nWh, wdd = wl / (nWw * nWh);
    for (int m = tid; m < NTOK; m += 64) {
      const int dd = m / 49, rem = m % 49, hh = rem / 7, w7 = rem % 7;
      const int gd = wdd * 4 + dd, gh = whh * 7 + hh, gw = www * 7 + w7;
      const int rd = gd < Dg - 4 ? 0 : (gd < Dg - 2 ? 1 : 2);
      const int rh = gh < Hg - 7 ? 0 : (gh < Hg - 3 ? 1 : 2);
      const int rw = gw < Wg - 7 ? 0 : (gw < Wg - 3 ? 1 : 2);
      cnt[m] = (rd * 3 + rh) * 3 + rw;
    }
  }
  for (int e = lane; e < 16 * NPAD; e += 32)  // zero padded P columns once
    Ps[wave][e / NPAD][e % NPAD] = (_Float16)0.f;
  __syncthreads();

  const int lm = lane & 15;
  const int hi = lane >> 4;

  for (int rt = wave; rt < 13; rt += 2) {
    v8f S[13];
#pragma unroll
    for (int ct = 0; ct < 13; ++ct) {
      v16h a, b;
#pragma unroll
      for (int i = 0; i < 16; ++i) {
        const int ka = (hi << 4) | i;
        a[i] = Qs[rt * 16 + lm][ka];
        b[i] = Ks[ct * 16 + lm][(hi << 4) | i];
      }
      v8f z = {};
      S[ct] = WMMA16(a, b, z);
    }
    int mrow[8];
#pragma unroll
    for (int r = 0; r < 8; ++r) mrow[r] = rt * 16 + r + 8 * hi;
    // relative position bias + shifted-window mask + padding mask
#pragma unroll
    for (int ct = 0; ct < 13; ++ct) {
      const int n = ct * 16 + lm;
      const bool nvalid = n < NTOK;
      int ddn = 0, hhn = 0, wwn = 0;
      if (nvalid) { ddn = n / 49; const int rem = n % 49; hhn = rem / 7; wwn = rem % 7; }
#pragma unroll
      for (int r = 0; r < 8; ++r) {
        const int m = mrow[r];
        float v = S[ct][r];
        if (nvalid && m < NTOK) {
          const int ddm = m / 49, rem = m % 49, hhm = rem / 7, wwm = rem % 7;
          const int ri = (ddm - ddn + 3) * 169 + (hhm - hhn + 6) * 13 + (wwm - wwn + 6);
          v += rpb[ri * heads + hI];
          if (shifted && cnt[m] != cnt[n]) v += -100.f;
        } else {
          v = -1e30f;
        }
        S[ct][r] = v;
      }
    }
    // softmax (row reduce: 13 tiles elementwise + 16-lane shuffle)
    float mx[8], sm[8];
#pragma unroll
    for (int r = 0; r < 8; ++r) {
      float m_ = -1e30f;
#pragma unroll
      for (int ct = 0; ct < 13; ++ct) m_ = fmaxf(m_, S[ct][r]);
#pragma unroll
      for (int o = 8; o >= 1; o >>= 1) m_ = fmaxf(m_, __shfl_xor(m_, o, 32));
      mx[r] = m_;
    }
#pragma unroll
    for (int r = 0; r < 8; ++r) {
      float s_ = 0.f;
#pragma unroll
      for (int ct = 0; ct < 13; ++ct) {
        const float e = __expf(S[ct][r] - mx[r]);
        S[ct][r] = e;
        s_ += e;
      }
#pragma unroll
      for (int o = 8; o >= 1; o >>= 1) s_ += __shfl_xor(s_, o, 32);
      sm[r] = s_;
    }
#pragma unroll
    for (int ct = 0; ct < 13; ++ct)
#pragma unroll
      for (int r = 0; r < 8; ++r)
        Ps[wave][r + 8 * hi][ct * 16 + lm] = (_Float16)(S[ct][r] / sm[r]);
    // O = P @ V  (K padded to 224 = 7 chunks of 32; pad cols of P are zero)
    v8f O0 = {}, O1 = {};
#pragma unroll
    for (int kc = 0; kc < 7; ++kc) {
      v16h a, b0, b1;
#pragma unroll
      for (int i = 0; i < 16; ++i) {
        const int ka = kc * 32 + ((hi << 4) | i);
        a[i] = Ps[wave][lm][ka];
        const int kb = kc * 32 + ((hi << 4) | i);
        b0[i] = Vs[kb][lm];
        b1[i] = Vs[kb][16 + lm];
      }
      O0 = WMMA16(a, b0, O0);
      O1 = WMMA16(a, b1, O1);
    }
#pragma unroll
    for (int r = 0; r < 8; ++r) {
      const int m = mrow[r];
      if (m < NTOK) {
        const size_t o = ((size_t)w * NTOK + m) * C + hI * 32;
        out[o + lm] = O0[r];
        out[o + 16 + lm] = O1[r];
      }
    }
  }
}

// ---------------------------------------------------------------- patch-merge gather (2,8,56,56,96)->(2,8,28,28,384)
__global__ void merge_gather_k(const float* __restrict__ X, float* __restrict__ Hc) {
  const size_t total = (size_t)2 * 8 * 28 * 28 * 384;
  for (size_t i = (size_t)blockIdx.x * blockDim.x + threadIdx.x; i < total;
       i += (size_t)gridDim.x * blockDim.x) {
    const int cc = (int)(i % 384);
    size_t e = i / 384;
    const int w2 = (int)(e % 28); e /= 28;
    const int h2 = (int)(e % 28); e /= 28;
    const int d = (int)(e % 8);
    const int b = (int)(e / 8);
    const int q = cc / 96, c = cc % 96;
    const int hsrc = 2 * h2 + (q & 1);
    const int wsrc = 2 * w2 + (q >> 1);
    Hc[i] = X[((((size_t)b * 8 + d) * 56 + hsrc) * 56 + wsrc) * 96 + c];
  }
}

// ---------------------------------------------------------------- global mean pool -> (2,192)
__global__ __launch_bounds__(192) void pool_k(const float* __restrict__ Hn, float* __restrict__ out) {
  const int b = blockIdx.x, c = threadIdx.x;
  const int T = 8 * 28 * 28;
  float s = 0.f;
  for (int t = 0; t < T; ++t) s += Hn[((size_t)b * T + t) * 192 + c];
  out[b * 192 + c] = s * (1.f / (float)T);
}

// ---------------------------------------------------------------- launcher
extern "C" void kernel_launch(void* const* d_in, const int* in_sizes, int n_in,
                              void* d_out, int out_size, void* d_ws, size_t ws_size,
                              hipStream_t stream) {
  (void)in_sizes; (void)n_in; (void)out_size; (void)ws_size;
  // Leaf order: [0]=x, then params leaves (JAX tree order: dict keys sorted, lists in order):
  // layers[0].blocks[0..1] (13 leaves each: fc1_b,fc1_w,fc2_b,fc2_w,n1_b,n1_g,n2_b,n2_g,
  //   proj_b,proj_w,qkv_b,qkv_w,rpb), merge{nb,ng,red_w}=27..29, layers[1].blocks at 30,43,
  // norm_b=56, norm_g=57, pe_b=58, pe_nb=59, pe_ng=60, pe_w=61.
  auto P = [&](int i) -> const float* { return (const float*)d_in[i]; };

  float* ws = (float*)d_ws;
  const size_t SZ = 4816896;  // 50176*96 == 12544*384
  float* X   = ws;            // current token grid
  float* H   = X + SZ;        // LN / merge scratch
  float* Wb  = H + SZ;        // window-ordered tokens / proj out
  float* A   = Wb + SZ;       // attention output (window order)
  float* QKV = A + SZ;        // 3*SZ
  float* MLP = QKV + 3 * SZ;  // T*4C

  patch_embed_k<<<50176, 96, 0, stream>>>((const float*)d_in[0], P(61), P(58), P(60), P(59), X);

  const int blkBase[2] = {1, 30};
  const int dims[2][3] = {{8, 56, 56}, {8, 28, 28}};
  const int Cs[2] = {96, 192};
  const int Hd[2] = {3, 6};

  for (int s = 0; s < 2; ++s) {
    const int Dg = dims[s][0], Hg = dims[s][1], Wg = dims[s][2];
    const int C = Cs[s], heads = Hd[s];
    const int Bn = 2 * (Dg / 4) * (Hg / 7) * (Wg / 7);
    const int T = Bn * 196;
    const size_t total = (size_t)T * C;
    for (int bi = 0; bi < 2; ++bi) {
      const int pb = blkBase[s] + bi * 13;
      const int sd = bi ? 2 : 0, sh = bi ? 3 : 0, sw = bi ? 3 : 0;
      layernorm_k<<<T, C, 0, stream>>>(X, H, P(pb + 5), P(pb + 4), C);
      win_partition_k<<<2048, 256, 0, stream>>>(H, Wb, Dg, Hg, Wg, C, sd, sh, sw, total);
      gemm_wmma_k<0, false><<<dim3((3 * C + 63) / 64, (T + 127) / 128), 256, 0, stream>>>(
          Wb, P(pb + 11), P(pb + 10), nullptr, QKV, T, 3 * C, C);
      win_attn_k<<<dim3(Bn, heads), 64, 0, stream>>>(QKV, P(pb + 12), A, heads, C, Dg, Hg, Wg, bi);
      gemm_wmma_k<0, false><<<dim3((C + 63) / 64, (T + 127) / 128), 256, 0, stream>>>(
          A, P(pb + 9), P(pb + 8), nullptr, Wb, T, C, C);
      win_reverse_add_k<<<2048, 256, 0, stream>>>(Wb, X, Dg, Hg, Wg, C, sd, sh, sw, total);
      layernorm_k<<<T, C, 0, stream>>>(X, H, P(pb + 7), P(pb + 6), C);
      gemm_wmma_k<1, false><<<dim3((4 * C + 63) / 64, (T + 127) / 128), 256, 0, stream>>>(
          H, P(pb + 1), P(pb + 0), nullptr, MLP, T, 4 * C, C);
      gemm_wmma_k<0, true><<<dim3((C + 63) / 64, (T + 127) / 128), 256, 0, stream>>>(
          MLP, P(pb + 3), P(pb + 2), X, X, T, C, 4 * C);
    }
    if (s == 0) {
      merge_gather_k<<<2048, 256, 0, stream>>>(X, H);
      layernorm_k<<<12544, 384, 0, stream>>>(H, H, P(28), P(27), 384);
      gemm_wmma_k<0, false><<<dim3((192 + 63) / 64, (12544 + 127) / 128), 256, 0, stream>>>(
          H, P(29), nullptr, nullptr, X, 12544, 192, 384);
    }
  }
  layernorm_k<<<12544, 192, 0, stream>>>(X, H, P(57), P(56), 192);
  pool_k<<<2, 192, 0, stream>>>(H, (float*)d_out);
}
